// SwinTransformerBlock_37615323578767
// MI455X (gfx1250) — matrix operand order repaired and checked by reading, not compiled, not run on hardware
//
#include <hip/hip_runtime.h>
#include <math.h>
#include <stdint.h>

typedef _Float16 f16;
typedef __attribute__((ext_vector_type(16))) _Float16 v16h;
typedef __attribute__((ext_vector_type(8)))  float    v8f;

#define C_CH   192
#define T_TOK  65536
#define NHEAD  6
#define HH     128
#define WWID   128
#define BATCH  4
#define PLANE  (BATCH * C_CH * HH * WWID)   // 12,582,912

// ---------------------------------------------------------------------------
// WMMA helpers (CDNA5 wave32, V_WMMA_F32_16X16X32_F16)
// ---------------------------------------------------------------------------
__device__ inline v8f wmma16(v16h a, v16h b, v8f c) {
  return __builtin_amdgcn_wmma_f32_16x16x32_f16(false, a, false, b, (short)0, c,
                                                false, false);
}
__device__ inline v16h neg16(v16h a) {
  v16h o;
#pragma unroll
  for (int e = 0; e < 16; ++e) o[e] = (_Float16)(-a[e]);
  return o;
}
// 16-bit A-matrix 16x32 layout: lane<16 holds M=lane, K={0..7,16..23};
// lane>=16 holds M=lane-16, K={8..15,24..31}.
__device__ inline v16h frag_a(const f16* base, int row0, int rs, int cs) {
  int l = threadIdx.x & 31;
  int half = l >> 4;
  int r = l & 15;
  v16h o;
#pragma unroll
  for (int e = 0; e < 16; ++e) {
    int k = e + half * 8 + ((e >= 8) ? 8 : 0);
    o[e] = base[(row0 + r) * rs + k * cs];
  }
  return o;
}
// 16-bit B-matrix 32x16: lane<16 holds N=lane, K={0..7,16..23};
// lane>=16 holds N=lane-16, K={8..15,24..31}.
__device__ inline v16h frag_b(const f16* base, int col0, int rs, int cs) {
  int l = threadIdx.x & 31;
  int half = l >> 4;
  int n = l & 15;
  v16h o;
#pragma unroll
  for (int e = 0; e < 16; ++e) {
    int k = e + half * 8 + ((e >= 8) ? 8 : 0);
    o[e] = base[k * rs + (col0 + n) * cs];
  }
  return o;
}
// window-token index -> linear [B,C,H,W] offset
__device__ inline int tok2lin(int t, int c) {
  int win = t >> 5, n = t & 31;
  int iw = win & 31, ih = (win >> 5) & 15, b = win >> 9;
  int h = ih * 8 + (n >> 2), w = iw * 4 + (n & 3);
  return ((b * C_CH + c) * HH + h) * WWID + w;
}

// ---------------------------------------------------------------------------
// CDNA5 async global->LDS (16B per lane, ASYNCcnt-tracked, no VGPR round-trip)
// LDS aperture keeps the LDS byte offset in flat addr[31:0].
// ---------------------------------------------------------------------------
__device__ inline void async_ld_b128(const void* gptr, const void* lptr) {
  uint64_t ga = (uint64_t)(uintptr_t)gptr;
  uint32_t la = (uint32_t)(uintptr_t)lptr;
  asm volatile("global_load_async_to_lds_b128 %0, %1, off"
               :: "v"(la), "v"(ga) : "memory");
}
__device__ inline void wait_async0() {
#if __has_builtin(__builtin_amdgcn_s_wait_asynccnt)
  __builtin_amdgcn_s_wait_asynccnt(0);
#else
  asm volatile("s_wait_asynccnt 0x0" ::: "memory");
#endif
}

// ---------------------------------------------------------------------------
// Packing / conversion kernels
// ---------------------------------------------------------------------------
__global__ void pack_x_kernel(const float* __restrict__ xr,
                              const float* __restrict__ xi,
                              f16* __restrict__ Xr, f16* __restrict__ Xi) {
  int idx = blockIdx.x * 256 + threadIdx.x;
  if (idx >= C_CH * T_TOK) return;
  int t = idx & (T_TOK - 1);
  int c = idx >> 16;
  int g = tok2lin(t, c);
  Xr[idx] = (f16)xr[g];
  Xi[idx] = (f16)xi[g];
}

__global__ void cvt16_kernel(const float* __restrict__ sr,
                             const float* __restrict__ si,
                             f16* __restrict__ dr, f16* __restrict__ di, int n) {
  int idx = blockIdx.x * 256 + threadIdx.x;
  if (idx >= n) return;
  dr[idx] = (f16)sr[idx];
  di[idx] = (f16)si[idx];
}

// ---------------------------------------------------------------------------
// Fused complex GEMM:  out[o,t] = sum_c W[o,c] * X[c,t]   (complex)
// modes: 0=QKV(store f16, scale q rows), 1=PROJ(+spatial residual,+BN stats),
//        2=MLP1(complex erf-GELU, store f16), 3=MLP2(+residual,+BN stats)
// Block: 256 threads = 8 waves; tile M=64 x N=64; each wave a 16x32 cplx strip
// (two 16x16 tiles sharing one A fragment) -> 8 WMMAs / 6 frag loads per K-step.
// ---------------------------------------------------------------------------
__global__ __launch_bounds__(256)
void cgemm_kernel(int M, int K, int mode,
                  const f16* __restrict__ Ar, const f16* __restrict__ Ai, // [M,K]
                  const f16* __restrict__ Br, const f16* __restrict__ Bi, // [K,T]
                  f16* __restrict__ Or16, f16* __restrict__ Oi16,         // [M,T]
                  float* __restrict__ Or32, float* __restrict__ Oi32,     // [M,T]
                  const float* __restrict__ resR,                          // [M,T]
                  const float* __restrict__ resI,
                  const float* __restrict__ xinR,                          // [B,C,H,W]
                  const float* __restrict__ xinI,
                  float* __restrict__ stats) {                             // [M*4]
  __shared__ f16 AsR[64 * 32], AsI[64 * 32];   // [m][k]
  __shared__ f16 BsR[32 * 64], BsI[32 * 64];   // [k][n]
  __shared__ float lstat[64 * 4];

  const int T = T_TOK;
  int t0 = blockIdx.x * 64;
  int m0 = blockIdx.y * 64;
  int tid = threadIdx.x;
  int wave = tid >> 5;
  int mw = wave >> 1;  // 0..3
  int nw = wave & 1;   // 0..1

  if (mode == 1 || mode == 3) lstat[tid] = 0.0f;

  // async staging indices (16B = 8 halves per transfer, one per thread/plane)
  int a_r = tid >> 2, a_s = (tid & 3) * 8;   // A: 64 rows x 4 segs
  int b_r = tid >> 3, b_s = (tid & 7) * 8;   // B: 32 rows x 8 segs

  v8f accR0 = {}, accI0 = {}, accR1 = {}, accI1 = {};

  for (int k0 = 0; k0 < K; k0 += 32) {
    __syncthreads();
    async_ld_b128(&Ar[(m0 + a_r) * K + k0 + a_s], &AsR[a_r * 32 + a_s]);
    async_ld_b128(&Ai[(m0 + a_r) * K + k0 + a_s], &AsI[a_r * 32 + a_s]);
    async_ld_b128(&Br[(k0 + b_r) * T + t0 + b_s], &BsR[b_r * 64 + b_s]);
    async_ld_b128(&Bi[(k0 + b_r) * T + t0 + b_s], &BsI[b_r * 64 + b_s]);
    wait_async0();
    __syncthreads();

    v16h ar = frag_a(AsR, mw * 16, 32, 1);
    v16h ai = frag_a(AsI, mw * 16, 32, 1);
    v16h ain = neg16(ai);
    v16h br0 = frag_b(BsR, nw * 32 + 0, 64, 1);
    v16h bi0 = frag_b(BsI, nw * 32 + 0, 64, 1);
    v16h br1 = frag_b(BsR, nw * 32 + 16, 64, 1);
    v16h bi1 = frag_b(BsI, nw * 32 + 16, 64, 1);
    accR0 = wmma16(ar, br0, accR0);
    accR0 = wmma16(ain, bi0, accR0);
    accI0 = wmma16(ar, bi0, accI0);
    accI0 = wmma16(ai, br0, accI0);
    accR1 = wmma16(ar, br1, accR1);
    accR1 = wmma16(ain, bi1, accR1);
    accI1 = wmma16(ar, bi1, accI1);
    accI1 = wmma16(ai, br1, accI1);
  }

  // ---- epilogue: D tile element e -> m = e + (l>>4)*8, n = l&15 ----
  int l = tid & 31;
  int half = l >> 4;
  int col = l & 15;
#pragma unroll
  for (int nt = 0; nt < 2; ++nt) {
    v8f zR = nt ? accR1 : accR0;
    v8f zI = nt ? accI1 : accI0;
    int nglob = t0 + nw * 32 + nt * 16 + col;
#pragma unroll
    for (int e = 0; e < 8; ++e) {
      int mloc = mw * 16 + e + half * 8;  // 0..63
      int mglob = m0 + mloc;
      float zr = zR[e];
      float zi = zI[e];
      if (mode == 0) {  // QKV: scale q rows by hd^-0.5
        float s = (mglob < C_CH) ? 0.17677669529663687f : 1.0f;
        Or16[mglob * T + nglob] = (f16)(zr * s);
        Oi16[mglob * T + nglob] = (f16)(zi * s);
      } else if (mode == 2) {  // complex exact-erf GELU
        float m = sqrtf(zr * zr + zi * zi);
        float g = 0.5f * m * (1.0f + erff(m * 0.70710678118654752f));
        float f = g / (m + 1e-8f);
        Or16[mglob * T + nglob] = (f16)(zr * f);
        Oi16[mglob * T + nglob] = (f16)(zi * f);
      } else {  // PROJ / MLP2: residual add + fp32 store + BN stats
        float rr, ri;
        if (mode == 1) {
          int g = tok2lin(nglob, mglob);
          rr = xinR[g];
          ri = xinI[g];
        } else {
          rr = resR[mglob * T + nglob];
          ri = resI[mglob * T + nglob];
        }
        float yr = rr + zr;
        float yi = ri + zi;
        Or32[mglob * T + nglob] = yr;
        Oi32[mglob * T + nglob] = yi;
        atomicAdd(&lstat[mloc * 4 + 0], yr);
        atomicAdd(&lstat[mloc * 4 + 1], yr * yr);
        atomicAdd(&lstat[mloc * 4 + 2], yi);
        atomicAdd(&lstat[mloc * 4 + 3], yi * yi);
      }
    }
  }
  if (mode == 1 || mode == 3) {
    __syncthreads();
    atomicAdd(&stats[(m0 + (tid >> 2)) * 4 + (tid & 3)], lstat[tid]);
  }
}

// ---------------------------------------------------------------------------
// Attention: one wave per (window, head). S = q^T conj(k) (+bias), magnitude
// softmax, O^T = attn * v^T. All 16x16x32 f16 WMMAs, fp32 softmax.
// ---------------------------------------------------------------------------
__global__ __launch_bounds__(32)
void attn_kernel(const f16* __restrict__ Qr, const f16* __restrict__ Qi, // QKV planes [576,T]
                 const float* __restrict__ rpb,                           // [105,6]
                 f16* __restrict__ AOr, f16* __restrict__ AOi) {          // [192,T]
  __shared__ f16 q_r[32 * 32], q_i[32 * 32];
  __shared__ f16 k_r[32 * 32], k_i[32 * 32];
  __shared__ f16 v_r[32 * 32], v_i[32 * 32];
  __shared__ float S_r[32 * 32], S_i[32 * 32];
  __shared__ f16 at_r[32 * 32], at_i[32 * 32];

  const int T = T_TOK;
  int win = blockIdx.x;
  int head = blockIdx.y;
  int t0 = win * 32;
  int l = threadIdx.x;

  int qrow = head * 32;            // q rows 0..191
  int krow = C_CH + head * 32;     // k rows 192..383
  int vrow = 2 * C_CH + head * 32; // v rows 384..575
  // async-stage q/k/v tiles: 128 x 16B per plane, 4 per lane
  for (int i = l; i < 128; i += 32) {
    int d = i >> 2, n = (i & 3) * 8;
    async_ld_b128(&Qr[(qrow + d) * T + t0 + n], &q_r[d * 32 + n]);
    async_ld_b128(&Qi[(qrow + d) * T + t0 + n], &q_i[d * 32 + n]);
    async_ld_b128(&Qr[(krow + d) * T + t0 + n], &k_r[d * 32 + n]);
    async_ld_b128(&Qi[(krow + d) * T + t0 + n], &k_i[d * 32 + n]);
    async_ld_b128(&Qr[(vrow + d) * T + t0 + n], &v_r[d * 32 + n]);
    async_ld_b128(&Qi[(vrow + d) * T + t0 + n], &v_i[d * 32 + n]);
  }
  wait_async0();
  __syncthreads();

  // S[n,m] = sum_d q[d,n]*conj(k[d,m]); q already scaled by hd^-0.5
#pragma unroll
  for (int nt = 0; nt < 2; ++nt) {
#pragma unroll
    for (int mt = 0; mt < 2; ++mt) {
      v8f sR = {};
      v8f sI = {};
      v16h aqr = frag_a(q_r, nt * 16, 1, 32);  // A[n,d] = q[d*32+n]
      v16h aqi = frag_a(q_i, nt * 16, 1, 32);
      v16h bkr = frag_b(k_r, mt * 16, 32, 1);  // B[d,m] = k[d*32+m]
      v16h bki = frag_b(k_i, mt * 16, 32, 1);
      sR = wmma16(aqr, bkr, sR);  // +qr*kr
      sR = wmma16(aqi, bki, sR);  // +qi*ki   (conj)
      sI = wmma16(aqi, bkr, sI);  // +qi*kr
      v16h aqrn = neg16(aqr);
      sI = wmma16(aqrn, bki, sI);  // -qr*ki
      int half = l >> 4, c = l & 15;
#pragma unroll
      for (int e = 0; e < 8; ++e) {
        int ns = nt * 16 + e + half * 8;
        int ms = mt * 16 + c;
        int r1 = ns >> 2, c1 = ns & 3, r2 = ms >> 2, c2 = ms & 3;
        int idx = (r1 - r2 + 7) * 7 + (c1 - c2 + 3);
        S_r[ns * 32 + ms] = sR[e] + rpb[idx * NHEAD + head];
        S_i[ns * 32 + ms] = sI[e];
      }
    }
  }
  __syncthreads();

  // magnitude softmax: lane l owns row l (recompute mag to keep VGPRs low)
  {
    float mx = -1e30f;
    for (int m = 0; m < 32; ++m) {
      float sr = S_r[l * 32 + m], si = S_i[l * 32 + m];
      float mg = sqrtf(sr * sr + si * si);
      mx = fmaxf(mx, mg);
    }
    float sum = 0.0f;
    for (int m = 0; m < 32; ++m) {
      float sr = S_r[l * 32 + m], si = S_i[l * 32 + m];
      float mg = sqrtf(sr * sr + si * si);
      sum += expf(mg - mx);
    }
    float inv = 1.0f / sum;
    for (int m = 0; m < 32; ++m) {
      float sr = S_r[l * 32 + m], si = S_i[l * 32 + m];
      float mg = sqrtf(sr * sr + si * si);
      float fac = expf(mg - mx) * inv / (mg + 1e-8f);
      at_r[l * 32 + m] = (f16)(sr * fac);
      at_i[l * 32 + m] = (f16)(si * fac);
    }
  }
  __syncthreads();

  // O^T[n,d] = sum_m attn[n,m] * v[d,m]
#pragma unroll
  for (int dt = 0; dt < 2; ++dt) {
#pragma unroll
    for (int nt = 0; nt < 2; ++nt) {
      v8f oR = {};
      v8f oI = {};
      v16h aar = frag_a(at_r, nt * 16, 32, 1);  // A[n,m]
      v16h aai = frag_a(at_i, nt * 16, 32, 1);
      v16h bvr = frag_b(v_r, dt * 16, 1, 32);   // B[m,d] = v[d*32+m]
      v16h bvi = frag_b(v_i, dt * 16, 1, 32);
      oR = wmma16(aar, bvr, oR);
      v16h aain = neg16(aai);
      oR = wmma16(aain, bvi, oR);
      oI = wmma16(aar, bvi, oI);
      oI = wmma16(aai, bvr, oI);
      int half = l >> 4, c = l & 15;
#pragma unroll
      for (int e = 0; e < 8; ++e) {
        int n = nt * 16 + e + half * 8;
        int d = dt * 16 + c;
        AOr[(head * 32 + d) * T + t0 + n] = (f16)oR[e];
        AOi[(head * 32 + d) * T + t0 + n] = (f16)oI[e];
      }
    }
  }
}

// ---------------------------------------------------------------------------
// BatchNorm finalize + apply
// ---------------------------------------------------------------------------
__global__ void bn_fin_kernel(const float* __restrict__ stats,
                              const float* __restrict__ gr, const float* __restrict__ br_,
                              const float* __restrict__ gi, const float* __restrict__ bi_,
                              float* __restrict__ bnp) {
  int c = blockIdx.x * 64 + threadIdx.x;
  if (c >= C_CH) return;
  const float inv = 1.0f / (float)T_TOK;
  float mr = stats[c * 4 + 0] * inv;
  float vr = stats[c * 4 + 1] * inv - mr * mr;
  float mi = stats[c * 4 + 2] * inv;
  float vi = stats[c * 4 + 3] * inv - mi * mi;
  float ar = gr[c] * rsqrtf(vr + 1e-5f);
  float ai = gi[c] * rsqrtf(vi + 1e-5f);
  bnp[c * 4 + 0] = ar;
  bnp[c * 4 + 1] = br_[c] - mr * ar;
  bnp[c * 4 + 2] = ai;
  bnp[c * 4 + 3] = bi_[c] - mi * ai;
}

__global__ void bn_apply_mid_kernel(const float* __restrict__ Yr,
                                    const float* __restrict__ Yi,
                                    const float* __restrict__ bnp,
                                    float* __restrict__ Xr32, float* __restrict__ Xi32,
                                    f16* __restrict__ Xr16, f16* __restrict__ Xi16) {
  int idx = blockIdx.x * 256 + threadIdx.x;
  if (idx >= C_CH * T_TOK) return;
  int c = idx >> 16;
  float vr = Yr[idx] * bnp[c * 4 + 0] + bnp[c * 4 + 1];
  float vi = Yi[idx] * bnp[c * 4 + 2] + bnp[c * 4 + 3];
  Xr32[idx] = vr;
  Xi32[idx] = vi;
  Xr16[idx] = (f16)vr;
  Xi16[idx] = (f16)vi;
}

__global__ void bn_apply_out_kernel(const float* __restrict__ Yr,
                                    const float* __restrict__ Yi,
                                    const float* __restrict__ bnp,
                                    float* __restrict__ out) {
  int idx = blockIdx.x * 256 + threadIdx.x;
  if (idx >= C_CH * T_TOK) return;
  int t = idx & (T_TOK - 1);
  int c = idx >> 16;
  float vr = Yr[idx] * bnp[c * 4 + 0] + bnp[c * 4 + 1];
  float vi = Yi[idx] * bnp[c * 4 + 2] + bnp[c * 4 + 3];
  int g = tok2lin(t, c);
  out[g] = vr;
  out[PLANE + g] = vi;
}

// ---------------------------------------------------------------------------
// Host launch
// ---------------------------------------------------------------------------
extern "C" void kernel_launch(void* const* d_in, const int* in_sizes, int n_in,
                              void* d_out, int out_size, void* d_ws, size_t ws_size,
                              hipStream_t stream) {
  (void)in_sizes; (void)n_in; (void)out_size; (void)ws_size;
  const float* x_real  = (const float*)d_in[0];
  const float* x_imag  = (const float*)d_in[1];
  const float* qkv_wr  = (const float*)d_in[2];
  const float* qkv_wi  = (const float*)d_in[3];
  const float* proj_wr = (const float*)d_in[4];
  const float* proj_wi = (const float*)d_in[5];
  const float* rpb     = (const float*)d_in[6];
  const float* n1_gr   = (const float*)d_in[7];
  const float* n1_br   = (const float*)d_in[8];
  const float* n1_gi   = (const float*)d_in[9];
  const float* n1_bi   = (const float*)d_in[10];
  const float* m1_wr   = (const float*)d_in[11];
  const float* m1_wi   = (const float*)d_in[12];
  const float* m2_wr   = (const float*)d_in[13];
  const float* m2_wi   = (const float*)d_in[14];
  const float* n2_gr   = (const float*)d_in[15];
  const float* n2_br   = (const float*)d_in[16];
  const float* n2_gi   = (const float*)d_in[17];
  const float* n2_bi   = (const float*)d_in[18];
  float* out = (float*)d_out;

  char* ws = (char*)d_ws;
  size_t off = 0;
  auto alloc = [&](size_t bytes) {
    size_t o = off;
    off += (bytes + 255) & ~(size_t)255;
    return o;
  };
  const size_t T = T_TOK;
  f16* wqkv_r = (f16*)(ws + alloc(576 * 192 * sizeof(f16)));
  f16* wqkv_i = (f16*)(ws + alloc(576 * 192 * sizeof(f16)));
  f16* wprj_r = (f16*)(ws + alloc(192 * 192 * sizeof(f16)));
  f16* wprj_i = (f16*)(ws + alloc(192 * 192 * sizeof(f16)));
  f16* wm1_r  = (f16*)(ws + alloc(768 * 192 * sizeof(f16)));
  f16* wm1_i  = (f16*)(ws + alloc(768 * 192 * sizeof(f16)));
  f16* wm2_r  = (f16*)(ws + alloc(192 * 768 * sizeof(f16)));
  f16* wm2_i  = (f16*)(ws + alloc(192 * 768 * sizeof(f16)));
  f16* Xr     = (f16*)(ws + alloc(192 * T * sizeof(f16)));
  f16* Xi     = (f16*)(ws + alloc(192 * T * sizeof(f16)));
  f16* QKVr   = (f16*)(ws + alloc(576 * T * sizeof(f16)));
  f16* QKVi   = (f16*)(ws + alloc(576 * T * sizeof(f16)));
  f16* AOr    = (f16*)(ws + alloc(192 * T * sizeof(f16)));
  f16* AOi    = (f16*)(ws + alloc(192 * T * sizeof(f16)));
  float* Y1r  = (float*)(ws + alloc(192 * T * sizeof(float)));
  float* Y1i  = (float*)(ws + alloc(192 * T * sizeof(float)));
  float* X1r32 = (float*)(ws + alloc(192 * T * sizeof(float)));
  float* X1i32 = (float*)(ws + alloc(192 * T * sizeof(float)));
  f16* X1r16  = (f16*)(ws + alloc(192 * T * sizeof(f16)));
  f16* X1i16  = (f16*)(ws + alloc(192 * T * sizeof(f16)));
  f16* Hr     = (f16*)(ws + alloc(768 * T * sizeof(f16)));
  f16* Hi     = (f16*)(ws + alloc(768 * T * sizeof(f16)));
  float* Y2r  = (float*)(ws + alloc(192 * T * sizeof(float)));
  float* Y2i  = (float*)(ws + alloc(192 * T * sizeof(float)));
  float* stats1 = (float*)(ws + alloc(192 * 4 * sizeof(float)));
  float* stats2 = (float*)(ws + alloc(192 * 4 * sizeof(float)));
  float* bn1  = (float*)(ws + alloc(192 * 4 * sizeof(float)));
  float* bn2  = (float*)(ws + alloc(192 * 4 * sizeof(float)));

  hipMemsetAsync(stats1, 0, 192 * 4 * sizeof(float), stream);
  hipMemsetAsync(stats2, 0, 192 * 4 * sizeof(float), stream);

  // weight conversions
  cvt16_kernel<<<(576 * 192 + 255) / 256, 256, 0, stream>>>(qkv_wr, qkv_wi, wqkv_r, wqkv_i, 576 * 192);
  cvt16_kernel<<<(192 * 192 + 255) / 256, 256, 0, stream>>>(proj_wr, proj_wi, wprj_r, wprj_i, 192 * 192);
  cvt16_kernel<<<(768 * 192 + 255) / 256, 256, 0, stream>>>(m1_wr, m1_wi, wm1_r, wm1_i, 768 * 192);
  cvt16_kernel<<<(192 * 768 + 255) / 256, 256, 0, stream>>>(m2_wr, m2_wi, wm2_r, wm2_i, 192 * 768);

  int nCT = C_CH * T_TOK;
  pack_x_kernel<<<(nCT + 255) / 256, 256, 0, stream>>>(x_real, x_imag, Xr, Xi);

  // QKV: [576,192] x [192,T]
  cgemm_kernel<<<dim3(T_TOK / 64, 576 / 64), 256, 0, stream>>>(
      576, 192, 0, wqkv_r, wqkv_i, Xr, Xi, QKVr, QKVi,
      nullptr, nullptr, nullptr, nullptr, nullptr, nullptr, nullptr);

  // attention: 2048 windows x 6 heads
  attn_kernel<<<dim3(2048, NHEAD), 32, 0, stream>>>(QKVr, QKVi, rpb, AOr, AOi);

  // proj + residual + BN1 stats
  cgemm_kernel<<<dim3(T_TOK / 64, 192 / 64), 256, 0, stream>>>(
      192, 192, 1, wprj_r, wprj_i, AOr, AOi, nullptr, nullptr,
      Y1r, Y1i, nullptr, nullptr, x_real, x_imag, stats1);

  bn_fin_kernel<<<3, 64, 0, stream>>>(stats1, n1_gr, n1_br, n1_gi, n1_bi, bn1);
  bn_apply_mid_kernel<<<(nCT + 255) / 256, 256, 0, stream>>>(Y1r, Y1i, bn1, X1r32, X1i32, X1r16, X1i16);

  // MLP1 + complex GELU
  cgemm_kernel<<<dim3(T_TOK / 64, 768 / 64), 256, 0, stream>>>(
      768, 192, 2, wm1_r, wm1_i, X1r16, X1i16, Hr, Hi,
      nullptr, nullptr, nullptr, nullptr, nullptr, nullptr, nullptr);

  // MLP2 + residual + BN2 stats
  cgemm_kernel<<<dim3(T_TOK / 64, 192 / 64), 256, 0, stream>>>(
      192, 768, 3, wm2_r, wm2_i, Hr, Hi, nullptr, nullptr,
      Y2r, Y2i, X1r32, X1i32, nullptr, nullptr, stats2);

  bn_fin_kernel<<<3, 64, 0, stream>>>(stats2, n2_gr, n2_br, n2_gi, n2_bi, bn2);
  bn_apply_out_kernel<<<(nCT + 255) / 256, 256, 0, stream>>>(Y2r, Y2i, bn2, out);
}
